// multihead_self_attention_56521769615696
// MI455X (gfx1250) — matrix operand order repaired and checked by reading, not compiled, not run on hardware
//
#include <hip/hip_runtime.h>

// ---------------------------------------------------------------------------
// MHA for MI455X (gfx1250): f32 -> f16 convert, WMMA f16 GEMMs + flash attn.
// GEMM: TDM (tensor_load_to_lds) double-buffered weight tiles in LDS,
//       software-pipelined A-fragment loads, v_wmma_f32_16x16x32_f16.
// ---------------------------------------------------------------------------

typedef __attribute__((ext_vector_type(16))) _Float16 v16h;
typedef __attribute__((ext_vector_type(8)))  _Float16 v8h;
typedef __attribute__((ext_vector_type(8)))  float    v8f;
typedef __attribute__((ext_vector_type(4)))  unsigned int u32x4;
typedef __attribute__((ext_vector_type(8)))  int          i32x8;
typedef __attribute__((ext_vector_type(4)))  int          i32x4;

#define B_  4
#define S_  2048
#define D_  1024
#define H_  16
#define DK_ 64

// ----- WMMA fragment loaders (contiguous k = 16h + s, same in A and B) ----------------
// A (16x32, MxK): lane l<16 -> row l, K 0..15 contiguous; lane l+16 -> K 16..31
__device__ __forceinline__ v16h load_a_frag(const _Float16* p, int ld) {
  int lane = threadIdx.x & 31;
  const _Float16* r = p + (size_t)(lane & 15) * ld + ((lane >> 4) << 4);
  return *(const v16h*)(r);
}

// B (32x16, KxN): lane n<16 -> col n, K 0..15 contiguous; lane n+16 -> K 16..31.
__device__ __forceinline__ v16h load_b_frag(const _Float16* p, int ld) {
  int lane = threadIdx.x & 31;
  const _Float16* r = p + (size_t)(lane & 15) * ld + ((lane >> 4) << 4);
  return *(const v16h*)(r);
}

__device__ __forceinline__ v8f wmma_f16(v16h a, v16h b, v8f c) {
  return __builtin_amdgcn_wmma_f32_16x16x32_f16(false, a, false, b, (short)0, c,
                                                false, false);
}

// ----- TDM: 2D f16 tile (tile_k x tile_n rows) global -> LDS ----------------
// Descriptor per CDNA5 ISA. Tensor dims == tile dims
// (no edge clipping needed; strides do the walk). data_size code 1 = 2 bytes.
// 6-arg builtin form (clang-23 / therock-10.0 toolchain).
__device__ __forceinline__ void tdm_load_2d_f16(unsigned lds_off,
                                                const _Float16* gptr,
                                                int tile_k, int tile_n,
                                                int row_stride_elems) {
  unsigned long long ga = (unsigned long long)(size_t)gptr;
  u32x4 g0;
  g0[0] = 1u;                                   // count=1 (valid), user desc
  g0[1] = lds_off;                              // LDS byte address
  g0[2] = (unsigned)(ga & 0xffffffffu);         // global_addr[31:0]
  g0[3] = (unsigned)((ga >> 32) & 0x01ffffffu)  // global_addr[56:32]
        | (2u << 30);                           // type = 2 ("image")
  i32x8 g1;
  g1[0] = (1 << 16);                            // wg_mask=0, data_size=2B
  g1[1] = (tile_k & 0xffff) << 16;              // tensor_dim0[15:0]
  g1[2] = ((tile_k >> 16) & 0xffff) | ((tile_n & 0xffff) << 16); // dim0 hi|dim1 lo
  g1[3] = ((tile_n >> 16) & 0xffff) | ((tile_k & 0xffff) << 16); // dim1 hi|tile_dim0
  g1[4] = (tile_n & 0xffff);                    // tile_dim1 (tile_dim2=0 -> 2D)
  g1[5] = row_stride_elems;                     // tensor_dim0_stride[31:0]
  g1[6] = 0;                                    // stride hi, dim1_stride lo
  g1[7] = 0;
  i32x4 z4 = {0, 0, 0, 0};
  i32x8 z8 = {0, 0, 0, 0, 0, 0, 0, 0};
  __builtin_amdgcn_tensor_load_to_lds(g0, g1, z4, z4, z8, 0);
}

// ----- f32 -> f16 conversion ------------------------------------------------
__global__ void cvt_f32_to_f16(const float* __restrict__ in,
                               _Float16* __restrict__ out, int n) {
  int i = blockIdx.x * blockDim.x + threadIdx.x;
  if (i < n) out[i] = (_Float16)in[i];
}

// ----- Generic GEMM:  Y[m][n] = sum_k X[m][k] * W[n][k]  --------------------
// mode 0: f16 head-split [b,h,s,dk] (Q,K)   mode 1: f16 [b,h,dk,s] (V)
// mode 2: f32 row-major (final projection)
// 256 thr (8 waves); block tile 256(M) x 64(N); wave tile 32 x 64.
// Weight tile (64n x 32k) staged in LDS by TDM, double buffered.
__global__ __launch_bounds__(256)
void gemm_xwT(const _Float16* __restrict__ X, const _Float16* __restrict__ W,
              _Float16* __restrict__ Yh, float* __restrict__ Yf, int mode) {
  __shared__ __align__(64) _Float16 Bs[2][64][32];   // 8 KiB, ping-pong

  const int wave = threadIdx.x >> 5;
  const int lane = threadIdx.x & 31;
  const int m0 = blockIdx.x * 256 + wave * 32;
  const int n0 = blockIdx.y * 64;
  const int NK = D_ / 32;

  const unsigned ldsoff[2] = {(unsigned)(size_t)&Bs[0][0][0],
                              (unsigned)(size_t)&Bs[1][0][0]};

  if (wave == 0)
    tdm_load_2d_f16(ldsoff[0], W + (size_t)n0 * D_, 32, 64, D_);

  v8f acc[2][4] = {};
  v16h a[2], an[2];
  a[0] = load_a_frag(X + (size_t)m0 * D_, D_);
  a[1] = load_a_frag(X + (size_t)(m0 + 16) * D_, D_);

  for (int kt = 0; kt < NK; ++kt) {
    const int cur = kt & 1;
    if (wave == 0) __builtin_amdgcn_s_wait_tensorcnt(0);  // Bs[cur] landed
    __syncthreads();   // publish Bs[cur]; everyone done reading Bs[cur^1]
    if (kt + 1 < NK) {
      if (wave == 0)
        tdm_load_2d_f16(ldsoff[cur ^ 1],
                        W + (size_t)n0 * D_ + (kt + 1) * 32, 32, 64, D_);
      an[0] = load_a_frag(X + (size_t)m0 * D_ + (kt + 1) * 32, D_);
      an[1] = load_a_frag(X + (size_t)(m0 + 16) * D_ + (kt + 1) * 32, D_);
    } else {
      an[0] = a[0]; an[1] = a[1];
    }
#pragma unroll
    for (int t = 0; t < 4; ++t) {
      v16h b = load_b_frag(&Bs[cur][t * 16][0], 32);
      acc[0][t] = wmma_f16(a[0], b, acc[0][t]);
      acc[1][t] = wmma_f16(a[1], b, acc[1][t]);
    }
    a[0] = an[0]; a[1] = an[1];
  }

  const int hl = lane >> 4, nl = lane & 15;
#pragma unroll
  for (int mi = 0; mi < 2; ++mi) {
    const int mrow0 = m0 + mi * 16;
    const int bidx = mrow0 / S_;
    const int srow = mrow0 % S_;    // 256-row blocks never straddle a batch
    if (mode == 0) {                // [b,h,s,dk] f16
#pragma unroll
      for (int t = 0; t < 4; ++t) {
        int n = n0 + t * 16 + nl;
        int h = n >> 6, dk = n & 63;
        _Float16* dst = Yh + ((size_t)bidx * H_ + h) * S_ * DK_ + dk;
#pragma unroll
        for (int r = 0; r < 8; ++r)
          dst[(size_t)(srow + r + 8 * hl) * DK_] = (_Float16)acc[mi][t][r];
      }
    } else if (mode == 1) {         // [b,h,dk,s] f16 (contiguous over r)
#pragma unroll
      for (int t = 0; t < 4; ++t) {
        int n = n0 + t * 16 + nl;
        int h = n >> 6, dk = n & 63;
        v8h pk;
#pragma unroll
        for (int r = 0; r < 8; ++r) pk[r] = (_Float16)acc[mi][t][r];
        int s = srow + 8 * hl;
        *(v8h*)(Yh + (((size_t)bidx * H_ + h) * DK_ + dk) * S_ + s) = pk;
      }
    } else {                        // f32 row-major
#pragma unroll
      for (int t = 0; t < 4; ++t) {
        int n = n0 + t * 16 + nl;
#pragma unroll
        for (int r = 0; r < 8; ++r)
          Yf[(size_t)(mrow0 + r + 8 * hl) * D_ + n] = acc[mi][t][r];
      }
    }
  }
}

// ----- Flash attention (causal, online softmax) -----------------------------
// grid = (S/128, B*H); 8 waves; each wave owns 16 query rows; key tiles of 32.
// K fragments for tile j+1 prefetched during softmax of tile j.
__global__ __launch_bounds__(256)
void flash_attn(const _Float16* __restrict__ Q, const _Float16* __restrict__ K,
                const _Float16* __restrict__ Vt, _Float16* __restrict__ O) {
  __shared__ __align__(32) _Float16 pbuf[8][16 * 32];  // per-wave P staging

  const int wave = threadIdx.x >> 5;
  const int lane = threadIdx.x & 31;
  const int hl = lane >> 4, nl = lane & 15;
  const int bh = blockIdx.y;
  const int b = bh / H_, h = bh % H_;
  const int q0 = blockIdx.x * 128 + wave * 16;

  const _Float16* Qb = Q + (size_t)bh * S_ * DK_;
  const _Float16* Kb = K + (size_t)bh * S_ * DK_;
  const _Float16* Vb = Vt + (size_t)bh * DK_ * S_;

  v16h qf0 = load_a_frag(Qb + (size_t)q0 * DK_ + 0, DK_);
  v16h qf1 = load_a_frag(Qb + (size_t)q0 * DK_ + 32, DK_);

  v8f oacc[4] = {};
  float mrow[8], lrow[8];
#pragma unroll
  for (int r = 0; r < 8; ++r) { mrow[r] = -1e30f; lrow[r] = 0.f; }

  const float sc = 0.125f;               // 1/sqrt(64)
  const int jtmax = (q0 + 15) >> 5;      // causal: last 32-wide key tile

  v16h kb[4];
  kb[0] = load_b_frag(Kb + 0, DK_);
  kb[1] = load_b_frag(Kb + 32, DK_);
  kb[2] = load_b_frag(Kb + (size_t)16 * DK_ + 0, DK_);
  kb[3] = load_b_frag(Kb + (size_t)16 * DK_ + 32, DK_);

  for (int jt = 0; jt <= jtmax; ++jt) {
    const int j0 = jt << 5;
    v8f s0 = {}, s1 = {};
    s0 = wmma_f16(qf0, kb[0], s0);
    s0 = wmma_f16(qf1, kb[1], s0);
    s1 = wmma_f16(qf0, kb[2], s1);
    s1 = wmma_f16(qf1, kb[3], s1);

    // prefetch next key tile + this tile's V fragments (overlap with softmax)
    v16h kn[4], vb[4];
    if (jt < jtmax) {
      const int jn = j0 + 32;
      kn[0] = load_b_frag(Kb + (size_t)jn * DK_ + 0, DK_);
      kn[1] = load_b_frag(Kb + (size_t)jn * DK_ + 32, DK_);
      kn[2] = load_b_frag(Kb + (size_t)(jn + 16) * DK_ + 0, DK_);
      kn[3] = load_b_frag(Kb + (size_t)(jn + 16) * DK_ + 32, DK_);
    } else {
      kn[0] = kb[0]; kn[1] = kb[1]; kn[2] = kb[2]; kn[3] = kb[3];
    }
#pragma unroll
    for (int t = 0; t < 4; ++t)
      vb[t] = load_b_frag(Vb + (size_t)(t * 16) * S_ + j0, S_);

    // online softmax per row (row r+8*hl lives across one 16-lane half)
#pragma unroll
    for (int r = 0; r < 8; ++r) {
      const int qi = q0 + 8 * hl + r;
      float a0 = (j0 + nl      <= qi) ? s0[r] * sc : -1e30f;
      float a1 = (j0 + 16 + nl <= qi) ? s1[r] * sc : -1e30f;
      float tmax = fmaxf(a0, a1);
#pragma unroll
      for (int off = 1; off < 16; off <<= 1)
        tmax = fmaxf(tmax, __shfl_xor(tmax, off, 32));
      float mnew = fmaxf(mrow[r], tmax);
      float cf = __expf(mrow[r] - mnew);
      mrow[r] = mnew;
      float p0 = __expf(a0 - mnew);
      float p1 = __expf(a1 - mnew);
      float ps = p0 + p1;
#pragma unroll
      for (int off = 1; off < 16; off <<= 1)
        ps += __shfl_xor(ps, off, 32);
      lrow[r] = lrow[r] * cf + ps;
#pragma unroll
      for (int t = 0; t < 4; ++t) oacc[t][r] *= cf;
      pbuf[wave][(r + 8 * hl) * 32 + nl]      = (_Float16)p0;
      pbuf[wave][(r + 8 * hl) * 32 + 16 + nl] = (_Float16)p1;
    }

    v16h pa = load_a_frag(&pbuf[wave][0], 32);
#pragma unroll
    for (int t = 0; t < 4; ++t)
      oacc[t] = wmma_f16(pa, vb[t], oacc[t]);

    kb[0] = kn[0]; kb[1] = kn[1]; kb[2] = kn[2]; kb[3] = kn[3];
  }

  _Float16* Ob = O + (size_t)b * S_ * D_ + h * 64;
#pragma unroll
  for (int t = 0; t < 4; ++t)
#pragma unroll
    for (int r = 0; r < 8; ++r) {
      int s = q0 + 8 * hl + r;
      Ob[(size_t)s * D_ + t * 16 + nl] = (_Float16)(oacc[t][r] / lrow[r]);
    }
}

// ---------------------------------------------------------------------------
extern "C" void kernel_launch(void* const* d_in, const int* in_sizes, int n_in,
                              void* d_out, int out_size, void* d_ws, size_t ws_size,
                              hipStream_t stream) {
  (void)in_sizes; (void)n_in; (void)out_size; (void)ws_size;
  const float* x  = (const float*)d_in[0];
  const float* wq = (const float*)d_in[1];
  const float* wk = (const float*)d_in[2];
  const float* wv = (const float*)d_in[3];
  const float* wo = (const float*)d_in[4];

  char* ws = (char*)d_ws;
  const size_t MiB = 1u << 20;
  _Float16* xh  = (_Float16*)(ws + 0 * MiB);   // 16 MiB  [B*S, D]
  _Float16* wqh = (_Float16*)(ws + 16 * MiB);  //  2 MiB  [D, D]
  _Float16* wkh = (_Float16*)(ws + 18 * MiB);
  _Float16* wvh = (_Float16*)(ws + 20 * MiB);
  _Float16* woh = (_Float16*)(ws + 22 * MiB);
  _Float16* qh  = (_Float16*)(ws + 24 * MiB);  // 16 MiB  [B,H,S,dk]
  _Float16* kh  = (_Float16*)(ws + 40 * MiB);  // 16 MiB  [B,H,S,dk]
  _Float16* vt  = (_Float16*)(ws + 56 * MiB);  // 16 MiB  [B,H,dk,S]
  _Float16* oh  = (_Float16*)(ws + 72 * MiB);  // 16 MiB  [B*S, D]

  const int NX = B_ * S_ * D_;
  const int NW = D_ * D_;
  cvt_f32_to_f16<<<(NX + 255) / 256, 256, 0, stream>>>(x,  xh,  NX);
  cvt_f32_to_f16<<<(NW + 255) / 256, 256, 0, stream>>>(wq, wqh, NW);
  cvt_f32_to_f16<<<(NW + 255) / 256, 256, 0, stream>>>(wk, wkh, NW);
  cvt_f32_to_f16<<<(NW + 255) / 256, 256, 0, stream>>>(wv, wvh, NW);
  cvt_f32_to_f16<<<(NW + 255) / 256, 256, 0, stream>>>(wo, woh, NW);

  dim3 gg(B_ * S_ / 256, D_ / 64);           // (32, 16)
  gemm_xwT<<<gg, 256, 0, stream>>>(xh, wqh, qh, nullptr, 0);
  gemm_xwT<<<gg, 256, 0, stream>>>(xh, wkh, kh, nullptr, 0);
  gemm_xwT<<<gg, 256, 0, stream>>>(xh, wvh, vt, nullptr, 1);

  dim3 ga(S_ / 128, B_ * H_);                // (16, 64)
  flash_attn<<<ga, 256, 0, stream>>>(qh, kh, vt, oh);

  gemm_xwT<<<gg, 256, 0, stream>>>(oh, woh, nullptr, (float*)d_out, 2);
}
